// CRF_84396107366758
// MI455X (gfx1250) — matrix operand order repaired and checked by reading, not compiled, not run on hardware
//
#include <hip/hip_runtime.h>

// CRF forward scan (B=32, T=512, N=64) via exp-domain WMMA on gfx1250.
//
// alpha_t[b,j] = scores[b,t,j] + m[b] + off[j] + log( sum_i exp(alpha[b,i]-m[b]) * E[i,j] )
// with E[i,j] = exp(trans[i,j] - off[j]),  off[j] = max_i trans[i,j]  (kept in VGPRs all T steps)
//
// - exp/log via native base-2 transcendentals (v_exp_f32 / v_log_f32)
// - LDS-only split barriers (s_wait_dscnt + s_barrier_signal/wait) so global output
//   stores drain in the background instead of stalling every step
// - scores software-pipelined one full step ahead

typedef __attribute__((ext_vector_type(16))) _Float16     v16h;
typedef __attribute__((ext_vector_type(8)))  float        v8f;
typedef __attribute__((ext_vector_type(4)))  unsigned int u32x4;

#define CRF_B 32
#define CRF_T 512
#define CRF_N 64

#define LOG2E 1.44269504088896340736f
#define LN2   0.69314718055994530942f

__device__ __forceinline__ float fast_exp(float x) {
    return __builtin_amdgcn_exp2f(x * LOG2E);      // v_exp_f32 (base-2)
}
__device__ __forceinline__ float fast_log(float x) {
    return __builtin_amdgcn_logf(x) * LN2;         // v_log_f32 (base-2)
}

// Workgroup barrier that only waits for LDS traffic (DScnt), NOT outstanding
// global stores (STOREcnt). Our inter-phase dependencies are LDS-only; the
// out[] stores are fire-and-forget and may complete across barriers.
__device__ __forceinline__ void barrier_lds_only() {
    asm volatile(
        "s_wait_dscnt 0x0\n\t"
        "s_barrier_signal -1\n\t"
        "s_barrier_wait -1"
        ::: "memory");
}

__global__ __launch_bounds__(128) void crf_fwd_wmma(
    const float* __restrict__ scores,      // (B, T, N)
    const float* __restrict__ trans,       // (N, N)
    float* __restrict__ out)               // (B, T, N)
{
    __shared__ __align__(16) float    alpha[16 * 64];   // current alphas (16 local batches)
    __shared__ __align__(16) _Float16 expA[16 * 64];    // exp(alpha - m) as f16, A operand
    __shared__ float mrow[16];                          // per-batch row max

    const int tid   = threadIdx.x;
    const int lane  = tid & 31;
    const int wave  = tid >> 5;            // 0..3 == N-tile index
    const int bbase = blockIdx.x * 16;     // 16 batches per block

    // ---------------- fixed B operand: E = exp(trans - colmax), f16, resident in VGPRs
    const int j     = wave * 16 + (lane & 15);   // output state column owned by this lane
    const int khalf = (lane >> 4) * 16;          // K rows held by this lane half

    float t0[16], t1[16];
    float cmax = -3.402823466e38f;
#pragma unroll
    for (int h = 0; h < 16; ++h) {
        t0[h] = trans[(khalf + h)      * CRF_N + j];   // K = khalf..khalf+15
        t1[h] = trans[(khalf + 32 + h) * CRF_N + j];   // K = khalf+32..khalf+47
        cmax  = fmaxf(cmax, fmaxf(t0[h], t1[h]));
    }
    // lanes L and L^16 cover complementary K halves of the same column
    cmax = fmaxf(cmax, __shfl_xor(cmax, 16, 32));
    const float offj = cmax;               // == -10000 exactly for the START column -> E = 1

    v16h bm0, bm1;
#pragma unroll
    for (int h = 0; h < 16; ++h) {
        bm0[h] = (_Float16)fast_exp(t0[h] - offj);
        bm1[h] = (_Float16)fast_exp(t1[h] - offj);
    }

    // ---------------- t = 0 : alpha0 = scores[:,0,:] + trans[START,:]
#pragma unroll
    for (int r = 0; r < 8; ++r) {
        int idx = tid * 8 + r;                       // 128 thr * 8 = 1024 = 16*64
        int b   = idx >> 6, jj = idx & 63;
        float v = scores[((size_t)(bbase + b) * CRF_T + 0) * CRF_N + jj] + trans[jj];
        alpha[idx] = v;
        out[((size_t)(bbase + b) * CRF_T + 0) * CRF_N + jj] = v;
    }

    const int bl = tid >> 3;          // batch handled in max/exp phase
    const int s0 = (tid & 7) * 8;     // state chunk base
    const int bp = lane & 15;         // A-matrix row (batch) for this lane
    const int q0 = lane >> 4;         // uint4 index offset into the expA row

    // preload scores for t = 1 (software pipeline, one step ahead)
    float sc[8];
#pragma unroll
    for (int vv = 0; vv < 8; ++vv) {
        int blo = vv + 8 * (lane >> 4);
        sc[vv] = scores[((size_t)(bbase + blo) * CRF_T + 1) * CRF_N + j];
    }

    barrier_lds_only();

    for (int t = 1; t < CRF_T; ++t) {
        // ---- phase 1: per-batch max (wave32 shfl over 8 lanes) + exp -> f16 LDS
        float vals[8];
        float mx = -3.402823466e38f;
#pragma unroll
        for (int k = 0; k < 8; ++k) {
            vals[k] = alpha[bl * 64 + s0 + k];
            mx = fmaxf(mx, vals[k]);
        }
#pragma unroll
        for (int d = 1; d < 8; d <<= 1) mx = fmaxf(mx, __shfl_xor(mx, d, 32));
        if ((tid & 7) == 0) mrow[bl] = mx;
#pragma unroll
        for (int k = 0; k < 8; ++k)
            expA[bl * 64 + s0 + k] = (_Float16)fast_exp(vals[k] - mx);
        barrier_lds_only();

        // ---- issue next step's score loads; a full step of latency to hide
        float scn[8];
        const int tn = (t + 1 < CRF_T) ? t + 1 : t;
#pragma unroll
        for (int vv = 0; vv < 8; ++vv) {
            int blo = vv + 8 * (lane >> 4);
            scn[vv] = scores[((size_t)(bbase + blo) * CRF_T + tn) * CRF_N + j];
        }

        // ---- phase 2: assemble A in contiguous k order k = 16*q0 + s, matching bm0/bm1 (16B-aligned LDS b128 loads)
        const u32x4* rowp = (const u32x4*)(expA + bp * 64);   // 64 halves = 8 uint4
        union { u32x4 q[2]; v16h v; } a0, a1;
        a0.q[0] = rowp[2 * q0];        // K = kb..kb+7        (kb = 16*q0)
        a0.q[1] = rowp[2 * q0 + 1];    // K = kb+8..kb+15
        a1.q[0] = rowp[2 * q0 + 4];    // K = 32+kb..32+kb+7
        a1.q[1] = rowp[2 * q0 + 5];    // K = 32+kb+8..32+kb+15

        v8f c = {};
        c = __builtin_amdgcn_wmma_f32_16x16x32_f16(false, a0.v, false, bm0,
                                                   (short)0, c, false, false);
        c = __builtin_amdgcn_wmma_f32_16x16x32_f16(false, a1.v, false, bm1,
                                                   (short)0, c, false, false);

        // ---- phase 3: log + offsets, write new alpha and output row
#pragma unroll
        for (int vv = 0; vv < 8; ++vv) {
            int blo = vv + 8 * (lane >> 4);         // D layout: VGPR vv -> batch blo
            float anew = sc[vv] + mrow[blo] + offj + fast_log(c[vv]);
            alpha[blo * 64 + j] = anew;
            out[((size_t)(bbase + blo) * CRF_T + t) * CRF_N + j] = anew;
            sc[vv] = scn[vv];
        }
        barrier_lds_only();
    }
}

extern "C" void kernel_launch(void* const* d_in, const int* in_sizes, int n_in,
                              void* d_out, int out_size, void* d_ws, size_t ws_size,
                              hipStream_t stream) {
    (void)in_sizes; (void)n_in; (void)out_size; (void)d_ws; (void)ws_size;
    const float* scores = (const float*)d_in[0];   // (32, 512, 64) f32
    const float* trans  = (const float*)d_in[1];   // (64, 64) f32
    float* out = (float*)d_out;                    // (32, 512, 64) f32

    // 2 blocks x 16 batches; 4 waves per block = 4 N-tiles of 16 states.
    crf_fwd_wmma<<<dim3(CRF_B / 16), dim3(128), 0, stream>>>(scores, trans, out);
}
